// LSTM_LM_42984032699039
// MI455X (gfx1250) — matrix operand order repaired and checked by reading, not compiled, not run on hardware
//
#include <hip/hip_runtime.h>

// ---------------------------------------------------------------------------
// LSTM LM on gfx1250: bf16 WMMA for all GEMMs (x-proj, recurrent, FC).
// Dims: V=32000 E=512 H=1024 L=2 B=16 T=256.
// ---------------------------------------------------------------------------

#define Vv 32000
#define Ee 512
#define Hh 1024
#define Bb 16
#define Tt 256
#define G4H (4 * Hh)    // 4096
#define Mrows (Tt * Bb) // 4096

typedef __attribute__((ext_vector_type(16))) __bf16 v16bf;
typedef __attribute__((ext_vector_type(8)))  float  v8f;

union FragBF {
    v16bf v;
    int4  q[2];
};

static __device__ __forceinline__ void load_frag(FragBF& f,
                                                 const unsigned short* p) {
    f.q[0] = *(const int4*)(p);
    f.q[1] = *(const int4*)(p + 16);
}

#define WMMA_BF16(d, av, bv_)                                                 \
    d = __builtin_amdgcn_wmma_f32_16x16x32_bf16(false, av, false, bv_,        \
                                                (short)0, d, false, false)

static __device__ __forceinline__ unsigned short f2bf(float f) {
    unsigned u = __builtin_bit_cast(unsigned, f);
    unsigned r = u + 0x7FFFu + ((u >> 16) & 1u); // round-to-nearest-even
    return (unsigned short)(r >> 16);
}

static __device__ __forceinline__ float sigmoidf_fast(float x) {
    return 1.0f / (1.0f + __expf(-x));
}

// ---------------------------------------------------------------------------
// f32 -> bf16 cast (weights)
// ---------------------------------------------------------------------------
__global__ void cast_bf16_kernel(const float* __restrict__ src,
                                 unsigned short* __restrict__ dst, size_t n) {
    size_t i = (size_t)blockIdx.x * blockDim.x + threadIdx.x;
    if (i < n) dst[i] = f2bf(src[i]);
}

// bias4 = bih + bhh
__global__ void bias_combine_kernel(const float* __restrict__ a,
                                    const float* __restrict__ b,
                                    float* __restrict__ out, int n) {
    int i = blockIdx.x * blockDim.x + threadIdx.x;
    if (i < n) out[i] = a[i] + b[i];
}

// ---------------------------------------------------------------------------
// Embedding gather -> time-major bf16 xs[T][B][E]
// ---------------------------------------------------------------------------
__global__ void embed_gather_kernel(const int* __restrict__ x,
                                    const float* __restrict__ emb,
                                    unsigned short* __restrict__ xs) {
    size_t i = (size_t)blockIdx.x * blockDim.x + threadIdx.x;
    if (i >= (size_t)Tt * Bb * Ee) return;
    int e = (int)(i & (Ee - 1));
    int b = (int)((i >> 9) & (Bb - 1));
    int t = (int)(i >> 13);
    int tok = x[b * Tt + t];
    xs[i] = f2bf(emb[(size_t)tok * Ee + e]);
}

// zero h ping buffer + sync counter before each scan
__global__ void scan_init_kernel(unsigned short* __restrict__ hbuf,
                                 unsigned int* __restrict__ sync) {
    int i = blockIdx.x * blockDim.x + threadIdx.x;
    if (i < Bb * Hh) hbuf[i] = 0;
    if (i == 0) *sync = 0u;
}

// ---------------------------------------------------------------------------
// GEMM: C[M,N](f32) = A[M,K](bf16 row-major) x W[N,K](bf16 row-major)^T + bias
// Wave tile: 32(M) x 64(N) = 8 accumulators; A frags reused across 4 N-tiles,
// B frags reused across 2 M-tiles -> 12 b128 loads per 8 WMMAs.
// K-loop is double-buffered (named buffers, constant indices, no scratch):
// WMMAs on buffer 0 overlap the loads of buffer 1 and vice versa.
// Block = 256 threads = 8 waves (2 M x 4 N) = 64 x 256 block tile.
// TRANS_BT: remap output row m = t*B+b  ->  row b*T+t (for the logits).
// K must be a multiple of 64 (512 / 1024 here).
// ---------------------------------------------------------------------------
template <bool TRANS_BT>
__global__ __launch_bounds__(256) void gemm_bf16_kernel(
        const unsigned short* __restrict__ A,
        const unsigned short* __restrict__ W,
        const float* __restrict__ bias,
        float* __restrict__ C, int K, int N) {
    const int wave = threadIdx.x >> 5;
    const int lane = threadIdx.x & 31;
    const int lrow = lane & 15;
    const int lhi  = lane >> 4;
    const int m0 = blockIdx.y * 64 + (wave & 1) * 32;
    const int n0 = blockIdx.x * 256 + (wave >> 1) * 64;

    const unsigned short* Ar[2];
    const unsigned short* Wr[4];
#pragma unroll
    for (int mi = 0; mi < 2; ++mi)
        Ar[mi] = A + (size_t)(m0 + mi * 16 + lrow) * K + lhi * 8;
#pragma unroll
    for (int nj = 0; nj < 4; ++nj)
        Wr[nj] = W + (size_t)(n0 + nj * 16 + lrow) * K + lhi * 8;

    v8f acc[2][4] = {};

    FragBF a0[2], b0[4], a1[2], b1[4];
    // prologue: buffer 0 <- k = 0
#pragma unroll
    for (int mi = 0; mi < 2; ++mi) load_frag(a0[mi], Ar[mi]);
#pragma unroll
    for (int nj = 0; nj < 4; ++nj) load_frag(b0[nj], Wr[nj]);

    for (int k = 0; k < K; k += 64) {
        // load buffer 1 <- k + 32
#pragma unroll
        for (int mi = 0; mi < 2; ++mi) load_frag(a1[mi], Ar[mi] + k + 32);
#pragma unroll
        for (int nj = 0; nj < 4; ++nj) load_frag(b1[nj], Wr[nj] + k + 32);
        // compute on buffer 0 (k)
#pragma unroll
        for (int mi = 0; mi < 2; ++mi)
#pragma unroll
            for (int nj = 0; nj < 4; ++nj)
                WMMA_BF16(acc[mi][nj], a0[mi].v, b0[nj].v);
        // load buffer 0 <- k + 64
        if (k + 64 < K) {
#pragma unroll
            for (int mi = 0; mi < 2; ++mi) load_frag(a0[mi], Ar[mi] + k + 64);
#pragma unroll
            for (int nj = 0; nj < 4; ++nj) load_frag(b0[nj], Wr[nj] + k + 64);
        }
        // compute on buffer 1 (k + 32)
#pragma unroll
        for (int mi = 0; mi < 2; ++mi)
#pragma unroll
            for (int nj = 0; nj < 4; ++nj)
                WMMA_BF16(acc[mi][nj], a1[mi].v, b1[nj].v);
    }

#pragma unroll
    for (int nj = 0; nj < 4; ++nj) {
        const int nn = n0 + nj * 16 + lrow;
        const float bv = bias ? bias[nn] : 0.0f;
#pragma unroll
        for (int mi = 0; mi < 2; ++mi)
#pragma unroll
            for (int r = 0; r < 8; ++r) {
                int mm = m0 + mi * 16 + lhi * 8 + r;
                size_t row = TRANS_BT
                                 ? ((size_t)(mm & (Bb - 1)) * Tt + (size_t)(mm >> 4))
                                 : (size_t)mm;
                C[row * (size_t)N + nn] = acc[mi][nj][r] + bv;
            }
    }
}

// ---------------------------------------------------------------------------
// LSTM recurrent scan (one layer). Persistent kernel: 64 blocks x 128 threads.
// Block owns hidden columns [hs, hs+16); wave g computes gate g's 16x16 tile
// (B=16 rows) with 32 WMMA k-steps per timestep. The full per-wave Whh tile
// is preloaded into 256 VGPRs (breg[32]) and stays register-resident for the
// whole 256-step scan; h fragments are double-buffered inside the unrolled
// K-loop. Gates exchanged through LDS; wave 0 keeps cell state c in registers.
// h double-buffered in global bf16; one atomic grid barrier per timestep.
// ---------------------------------------------------------------------------
__global__ __launch_bounds__(128) void lstm_scan_kernel(
        const float* __restrict__ xp,              // [T*B, 4H] pre-activations
        const unsigned short* __restrict__ Whh,    // [4H, H] bf16
        unsigned short* __restrict__ hbuf,         // [2][B][H] bf16 (buf0 zeroed)
        unsigned short* __restrict__ ys,           // [T*B, H] bf16 output
        float* __restrict__ hn, float* __restrict__ cn, // [B*H] finals
        unsigned int* __restrict__ sync) {
    const int tid  = threadIdx.x;
    const int gate = tid >> 5;     // 0=i 1=f 2=g 3=o
    const int lane = tid & 31;
    const int lrow = lane & 15;
    const int lhi  = lane >> 4;
    const int hs   = blockIdx.x * 16;

    __shared__ float gbuf[4][Bb][16];

    v8f cst = {}; // cell state (only wave 0's copy is meaningful)

    const unsigned short* Wrow =
        Whh + (size_t)(gate * Hh + hs + lrow) * Hh + lhi * 8;

    // Whh tile -> registers for the entire scan (256 VGPRs/wave)
    FragBF breg[32];
#pragma unroll
    for (int kk = 0; kk < 32; ++kk) load_frag(breg[kk], Wrow + kk * 32);

    for (int t = 0; t < Tt; ++t) {
        const unsigned short* hread  = hbuf + (size_t)(t & 1) * Bb * Hh;
        unsigned short*       hwrite = hbuf + (size_t)((t + 1) & 1) * Bb * Hh;

        const unsigned short* Arow = hread + (size_t)lrow * Hh + lhi * 8;
        v8f acc = {};
        FragBF ha, hb;
        load_frag(ha, Arow);
#pragma unroll
        for (int kk = 0; kk < 32; kk += 2) {
            load_frag(hb, Arow + (kk + 1) * 32);
            WMMA_BF16(acc, ha.v, breg[kk].v);
            if (kk + 2 < 32) load_frag(ha, Arow + (kk + 2) * 32);
            WMMA_BF16(acc, hb.v, breg[kk + 1].v);
        }

        // add x-projection, activate, publish to LDS
#pragma unroll
        for (int r = 0; r < 8; ++r) {
            int bb = lhi * 8 + r;
            float v = acc[r] +
                xp[(size_t)(t * Bb + bb) * G4H + gate * Hh + hs + lrow];
            v = (gate == 2) ? tanhf(v) : sigmoidf_fast(v);
            gbuf[gate][bb][lrow] = v;
        }
        __syncthreads();

        if (gate == 0) { // wave 0 combines gates, owns c, writes h
#pragma unroll
            for (int r = 0; r < 8; ++r) {
                int bb = lhi * 8 + r;
                float iv = gbuf[0][bb][lrow];
                float fv = gbuf[1][bb][lrow];
                float gv = gbuf[2][bb][lrow];
                float ov = gbuf[3][bb][lrow];
                float cv = fv * cst[r] + iv * gv;
                cst[r] = cv;
                float hv = ov * tanhf(cv);
                unsigned short hbv = f2bf(hv);
                hwrite[(size_t)bb * Hh + hs + lrow] = hbv;
                ys[(size_t)(t * Bb + bb) * Hh + hs + lrow] = hbv;
                if (t == Tt - 1) {
                    hn[bb * Hh + hs + lrow] = hv;
                    cn[bb * Hh + hs + lrow] = cv;
                }
            }
        }

        // ------- grid barrier: make new h visible to all 64 blocks -------
        __threadfence();
        __syncthreads();
        if (tid == 0) {
            atomicAdd(sync, 1u);
            unsigned target = 64u * (unsigned)(t + 1);
            volatile unsigned* vs = sync;
            while (*vs < target) { __builtin_amdgcn_s_sleep(2); }
        }
        __syncthreads();
        __threadfence();
    }
}

// ---------------------------------------------------------------------------
// Host launch
// ---------------------------------------------------------------------------
extern "C" void kernel_launch(void* const* d_in, const int* in_sizes, int n_in,
                              void* d_out, int out_size, void* d_ws, size_t ws_size,
                              hipStream_t stream) {
    (void)in_sizes; (void)n_in; (void)out_size; (void)ws_size;

    const int*   x    = (const int*)d_in[0];
    const float* emb  = (const float*)d_in[1];
    const float* Wih0 = (const float*)d_in[2];
    const float* Whh0 = (const float*)d_in[3];
    const float* bih0 = (const float*)d_in[4];
    const float* bhh0 = (const float*)d_in[5];
    const float* Wih1 = (const float*)d_in[6];
    const float* Whh1 = (const float*)d_in[7];
    const float* bih1 = (const float*)d_in[8];
    const float* bhh1 = (const float*)d_in[9];
    const float* fcW  = (const float*)d_in[10];
    const float* fcb  = (const float*)d_in[11];

    float* out = (float*)d_out;
    float* logits = out;                                   // [B,T,V]
    float* h_n = out + (size_t)Bb * Tt * Vv;               // [2,B,H]
    float* c_n = h_n + (size_t)2 * Bb * Hh;                // [2,B,H]

    // ---- workspace carve-up ----
    char* p = (char*)d_ws;
    auto carve = [&](size_t bytes) {
        char* r = p;
        p += (bytes + 255) & ~(size_t)255;
        return r;
    };
    unsigned short* wih0b = (unsigned short*)carve((size_t)G4H * Ee * 2);
    unsigned short* whh0b = (unsigned short*)carve((size_t)G4H * Hh * 2);
    unsigned short* wih1b = (unsigned short*)carve((size_t)G4H * Hh * 2);
    unsigned short* whh1b = (unsigned short*)carve((size_t)G4H * Hh * 2);
    unsigned short* fcwb  = (unsigned short*)carve((size_t)Vv * Hh * 2);
    unsigned short* xs    = (unsigned short*)carve((size_t)Mrows * Ee * 2);
    unsigned short* ys0   = (unsigned short*)carve((size_t)Mrows * Hh * 2);
    unsigned short* ys1   = (unsigned short*)carve((size_t)Mrows * Hh * 2);
    float*          xp    = (float*)carve((size_t)Mrows * G4H * 4);
    float*          bias0 = (float*)carve((size_t)G4H * 4);
    float*          bias1 = (float*)carve((size_t)G4H * 4);
    unsigned short* hbuf  = (unsigned short*)carve((size_t)2 * Bb * Hh * 2);
    unsigned int*   sync  = (unsigned int*)carve(256);

    auto cast = [&](const float* s, unsigned short* d, size_t n) {
        cast_bf16_kernel<<<(unsigned)((n + 255) / 256), 256, 0, stream>>>(s, d, n);
    };
    cast(Wih0, wih0b, (size_t)G4H * Ee);
    cast(Whh0, whh0b, (size_t)G4H * Hh);
    cast(Wih1, wih1b, (size_t)G4H * Hh);
    cast(Whh1, whh1b, (size_t)G4H * Hh);
    cast(fcW,  fcwb,  (size_t)Vv * Hh);

    bias_combine_kernel<<<G4H / 256, 256, 0, stream>>>(bih0, bhh0, bias0, G4H);
    bias_combine_kernel<<<G4H / 256, 256, 0, stream>>>(bih1, bhh1, bias1, G4H);

    {
        size_t n = (size_t)Tt * Bb * Ee;
        embed_gather_kernel<<<(unsigned)((n + 255) / 256), 256, 0, stream>>>(x, emb, xs);
    }

    dim3 blk(256);
    // ---- layer 0: x-proj then scan ----
    gemm_bf16_kernel<false><<<dim3(G4H / 256, Mrows / 64), blk, 0, stream>>>(
        xs, wih0b, bias0, xp, Ee, G4H);
    scan_init_kernel<<<(Bb * Hh + 255) / 256, 256, 0, stream>>>(hbuf, sync);
    lstm_scan_kernel<<<64, 128, 0, stream>>>(xp, whh0b, hbuf, ys0,
                                             h_n, c_n, sync);
    // ---- layer 1 ----
    gemm_bf16_kernel<false><<<dim3(G4H / 256, Mrows / 64), blk, 0, stream>>>(
        ys0, wih1b, bias1, xp, Hh, G4H);
    scan_init_kernel<<<(Bb * Hh + 255) / 256, 256, 0, stream>>>(hbuf, sync);
    lstm_scan_kernel<<<64, 128, 0, stream>>>(xp, whh1b, hbuf, ys1,
                                             h_n + (size_t)Bb * Hh,
                                             c_n + (size_t)Bb * Hh, sync);
    // ---- output projection: logits[b,t,:] ----
    gemm_bf16_kernel<true><<<dim3(Vv / 256, Mrows / 64), blk, 0, stream>>>(
        ys1, fcwb, fcb, logits, Hh, Vv);
}
